// MoENetwork_43499428774597
// MI455X (gfx1250) — compile-verified, matrix operand order checked by reading, not run on hardware
//
#include <hip/hip_runtime.h>
#include <hip/hip_bf16.h>
#include <stdint.h>

#define TOKENS 1024
#define IN_W   512
#define HID    256
#define OUT_W  512
#define NEXP   64
#define TILE_M 16

typedef __attribute__((ext_vector_type(16))) __bf16 v16bf;
typedef __attribute__((ext_vector_type(4)))  __bf16 v4bf;
typedef __attribute__((ext_vector_type(8)))  float  v8f;

__device__ __forceinline__ float softplus_f(float v) {
  return (v > 20.f) ? v : log1pf(__expf(v));
}

// ---------------------------------------------------------------------------
// Zero output accumulator + per-expert counts
// ---------------------------------------------------------------------------
__global__ void moe_zero(float* __restrict__ out, int* __restrict__ counts) {
  int i = blockIdx.x * blockDim.x + threadIdx.x;
  if (i < TOKENS * OUT_W) out[i] = 0.f;
  if (i < NEXP) counts[i] = 0;
}

// ---------------------------------------------------------------------------
// Router: one wave (32 lanes) per token; 2 experts per lane (64 experts).
// h = x@mixer + noise * softplus(x@noise_controller); wave-wide top-2 + softmax.
// ---------------------------------------------------------------------------
__global__ __launch_bounds__(256) void moe_router(
    const float* __restrict__ x, const float* __restrict__ noise,
    const float* __restrict__ mixer, const float* __restrict__ ncont,
    int* __restrict__ ridx, float* __restrict__ rsc)
{
  const int wave = threadIdx.x >> 5;
  const int lane = threadIdx.x & 31;
  const int t = blockIdx.x * 8 + wave;          // grid = TOKENS/8 -> t < TOKENS
  const float* xr = x + (size_t)t * IN_W;

  float a0 = 0.f, a1 = 0.f, c0 = 0.f, c1 = 0.f;
  for (int i = 0; i < IN_W; ++i) {
    const float xv = xr[i];
    a0 = fmaf(xv, mixer[i * NEXP + lane],      a0);
    a1 = fmaf(xv, mixer[i * NEXP + lane + 32], a1);
    c0 = fmaf(xv, ncont[i * NEXP + lane],      c0);
    c1 = fmaf(xv, ncont[i * NEXP + lane + 32], c1);
  }
  const float h0 = a0 + noise[t * NEXP + lane]      * softplus_f(c0);
  const float h1 = a1 + noise[t * NEXP + lane + 32] * softplus_f(c1);

  float v1, v2; int i1, i2;
  if (h0 >= h1) { v1 = h0; i1 = lane;      v2 = h1; i2 = lane + 32; }
  else          { v1 = h1; i1 = lane + 32; v2 = h0; i2 = lane;      }

  // butterfly top-2 merge across the wave (wave32)
  for (int off = 16; off >= 1; off >>= 1) {
    const float ov1 = __shfl_xor(v1, off, 32);
    const float ov2 = __shfl_xor(v2, off, 32);
    const int   oi1 = __shfl_xor(i1, off, 32);
    const int   oi2 = __shfl_xor(i2, off, 32);
    if (ov1 > v1) {
      if (v1 >= ov2) { v2 = v1; i2 = i1; } else { v2 = ov2; i2 = oi2; }
      v1 = ov1; i1 = oi1;
    } else if (ov1 > v2) { v2 = ov1; i2 = oi1; }
  }

  if (lane == 0) {
    const float e2  = __expf(v2 - v1);
    const float inv = 1.f / (1.f + e2);
    ridx[t * 2 + 0] = i1;  ridx[t * 2 + 1] = i2;
    rsc[t * 2 + 0]  = inv; rsc[t * 2 + 1]  = e2 * inv;
  }
}

// ---------------------------------------------------------------------------
// Scatter (token,k-slot) assignments into per-expert buckets
// ---------------------------------------------------------------------------
__global__ void moe_scatter(const int* __restrict__ ridx, int* __restrict__ counts,
                            int* __restrict__ buckets) {
  int i = blockIdx.x * blockDim.x + threadIdx.x;     // 0 .. 2*TOKENS-1
  if (i < TOKENS * 2) {
    const int ex   = ridx[i];
    const int slot = atomicAdd(&counts[ex], 1);
    buckets[ex * TOKENS + slot] = i;                 // packed: token = i>>1, k = i&1
  }
}

// ---------------------------------------------------------------------------
// Stage one 32 x W f32 weight slab from global into LDS, transposed to
// B-fragment order ([n][k], 32 k contiguous per column), converting to bf16.
// Global side: one coalesced b128 per thread-iteration.
// ---------------------------------------------------------------------------
template <int W>
__device__ __forceinline__ void stage_slab(const float* __restrict__ src,
                                           __bf16* __restrict__ dst, int tid) {
  constexpr int V4_PER_ROW = W / 4;
#pragma unroll
  for (int v4i = tid; v4i < 32 * V4_PER_ROW; v4i += 256) {
    const int kk = v4i / V4_PER_ROW;            // power-of-two -> shift
    const int cc = (v4i % V4_PER_ROW) * 4;
    const float4 f = *(const float4*)&src[(size_t)kk * W + cc];
    __bf16* d = &dst[cc * 32 + kk];
    d[0]      = (__bf16)f.x;
    d[32]     = (__bf16)f.y;
    d[64]     = (__bf16)f.z;
    d[96]     = (__bf16)f.w;
  }
}

// ---------------------------------------------------------------------------
// Expert FFN: one block per expert, 8 waves. bf16 WMMA, f32 accumulate.
//   - X tile gathered via GLOBAL_LOAD_ASYNC_TO_LDS_B128 (ASYNCcnt), then bf16.
//   - W slabs staged transposed ([n][k]) so B fragments are contiguous 32B
//     v16bf LDS loads (same shape as A fragments) -> no scalar gathers.
//   - Double-buffered slabs: one barrier per k-step, loads overlap WMMA.
// ---------------------------------------------------------------------------
__global__ __launch_bounds__(256) void moe_expert_ffn(
    const float* __restrict__ x,
    const float* __restrict__ w1s, const float* __restrict__ b1s,
    const float* __restrict__ w2s, const float* __restrict__ b2s,
    const int* __restrict__ counts, const int* __restrict__ buckets,
    const float* __restrict__ rsc, float* __restrict__ out)
{
  const int e     = blockIdx.x;
  const int tid   = threadIdx.x;
  const int wave  = tid >> 5;
  const int lane  = tid & 31;
  const int lrow  = lane & 15;   // M (for A) / N (for B, C/D) sub-index
  const int lhalf = lane >> 4;   // K-half selector per WMMA VGPR layout

  const float* __restrict__ W1 = w1s + (size_t)e * IN_W * HID;
  const float* __restrict__ W2 = w2s + (size_t)e * HID * OUT_W;
  const float* __restrict__ B1 = b1s + e * HID;
  const float* __restrict__ B2 = b2s + e * OUT_W;

  const int n = counts[e];

  __shared__ __bf16 sX[TILE_M][IN_W];        // 16 KB: token tile (bf16)
  __shared__ __bf16 sH[TILE_M][HID];         //  8 KB: hidden activations (bf16)
  __shared__ union {
    float  xf[TILE_M * IN_W];                // 32 KB: f32 async staging of X
    __bf16 w[2][32 * OUT_W];                 // 2x32 KB: transposed W slabs
  } sU;                                      // 64 KB
  __shared__ int    sTok[TILE_M];
  __shared__ float  sScore[TILE_M];

  const unsigned xfBase = (unsigned)(uintptr_t)&sU.xf[0];

  for (int t0 = 0; t0 < n; t0 += TILE_M) {
    if (tid < TILE_M) {
      const int slot   = t0 + tid;
      const int cslot  = (slot < n) ? slot : (n - 1);
      const int packed = buckets[e * TOKENS + cslot];
      sTok[tid]   = packed >> 1;
      sScore[tid] = (slot < n) ? rsc[(packed >> 1) * 2 + (packed & 1)] : 0.f;
    }
    __syncthreads();

    // ---- async gather of the 16 x 512 f32 X tile into LDS (ASYNCcnt path)
    for (int v4i = tid; v4i < TILE_M * IN_W / 4; v4i += 256) {
      const int r = v4i >> 7;                 // 128 vec4 per row
      const int c = (v4i & 127) << 2;
      const float* g = &x[(size_t)sTok[r] * IN_W + c];
      const unsigned l = xfBase + (unsigned)((r * IN_W + c) * 4);
      asm volatile("global_load_async_to_lds_b128 %0, %1, off"
                   :: "v"(l), "v"(g) : "memory");
    }
    asm volatile("s_wait_asynccnt 0x0" ::: "memory");
    __syncthreads();

    // ---- vectorized f32 -> bf16 convert into the WMMA A-operand tile
    for (int v4i = tid; v4i < TILE_M * IN_W / 4; v4i += 256) {
      const int r = v4i >> 7;
      const int c = (v4i & 127) << 2;
      const float4 f = *(const float4*)&sU.xf[r * IN_W + c];
      const v4bf o = {(__bf16)f.x, (__bf16)f.y, (__bf16)f.z, (__bf16)f.w};
      *(v4bf*)&sX[r][c] = o;
    }
    __syncthreads();   // xf dead from here; sU.w may be written

    // ---------------- layer 1: [16x512] @ [512x256] -> relu -> sH ----------
    v8f acc1[2] = {};
    if (lane == 0)     // warm L2 for layer-2 weights while layer 1 runs
      __builtin_prefetch(W2 + wave * 1024, 0, 1);
    stage_slab<HID>(W1, sU.w[0], tid);                     // prologue slab k0=0
    __syncthreads();
    int p = 0;
    for (int k0 = 0; k0 < IN_W; k0 += 32) {
      if (k0 + 32 < IN_W)                                  // stage next slab
        stage_slab<HID>(&W1[(size_t)(k0 + 32) * HID], sU.w[p ^ 1], tid);
      const v16bf a = *(const v16bf*)&sX[lrow][k0 + lhalf * 16];
#pragma unroll
      for (int ct = 0; ct < 2; ++ct) {
        const int col = wave * 32 + ct * 16 + lrow;
        const v16bf b = *(const v16bf*)&sU.w[p][col * 32 + lhalf * 16];
        acc1[ct] = __builtin_amdgcn_wmma_f32_16x16x32_bf16(
            false, a, false, b, (short)0, acc1[ct], false, false);
      }
      __syncthreads();
      p ^= 1;
    }

    // bias + relu -> sH
#pragma unroll
    for (int ct = 0; ct < 2; ++ct) {
      const int col  = wave * 32 + ct * 16 + lrow;
      const float bs = B1[col];
#pragma unroll
      for (int g = 0; g < 8; ++g) {
        const int m = g + lhalf * 8;
        const float v = acc1[ct][g] + bs;
        sH[m][col] = (__bf16)(v > 0.f ? v : 0.f);
      }
    }
    __syncthreads();

    // ---------------- layer 2: [16x256] @ [256x512] -> out ----------------
    v8f acc2[4] = {};
    stage_slab<OUT_W>(W2, sU.w[0], tid);                   // prologue slab k0=0
    __syncthreads();
    p = 0;
    for (int k0 = 0; k0 < HID; k0 += 32) {
      if (k0 + 32 < HID)                                   // stage next slab
        stage_slab<OUT_W>(&W2[(size_t)(k0 + 32) * OUT_W], sU.w[p ^ 1], tid);
      const v16bf a = *(const v16bf*)&sH[lrow][k0 + lhalf * 16];
#pragma unroll
      for (int ct = 0; ct < 4; ++ct) {
        const int col = wave * 64 + ct * 16 + lrow;
        const v16bf b = *(const v16bf*)&sU.w[p][col * 32 + lhalf * 16];
        acc2[ct] = __builtin_amdgcn_wmma_f32_16x16x32_bf16(
            false, a, false, b, (short)0, acc2[ct], false, false);
      }
      __syncthreads();
      p ^= 1;
    }

    // bias + score scale + atomic accumulate into out
#pragma unroll
    for (int ct = 0; ct < 4; ++ct) {
      const int col  = wave * 64 + ct * 16 + lrow;
      const float bs = B2[col];
#pragma unroll
      for (int g = 0; g < 8; ++g) {
        const int m = g + lhalf * 8;
        if (t0 + m < n) {
          const float v = (acc2[ct][g] + bs) * sScore[m];
          atomicAdd(&out[(size_t)sTok[m] * OUT_W + col], v);
        }
      }
    }
    __syncthreads();
  }
}

// ---------------------------------------------------------------------------
extern "C" void kernel_launch(void* const* d_in, const int* in_sizes, int n_in,
                              void* d_out, int out_size, void* d_ws, size_t ws_size,
                              hipStream_t stream) {
  const float* x     = (const float*)d_in[0];
  const float* noise = (const float*)d_in[1];
  const float* w1s   = (const float*)d_in[2];
  const float* b1s   = (const float*)d_in[3];
  const float* w2s   = (const float*)d_in[4];
  const float* b2s   = (const float*)d_in[5];
  const float* mixer = (const float*)d_in[6];
  const float* ncont = (const float*)d_in[7];
  float* out = (float*)d_out;

  // workspace layout (~272 KB total)
  char*  ws      = (char*)d_ws;
  int*   counts  = (int*)ws;                    // 64 * 4
  int*   ridx    = (int*)(ws + 256);            // 2048 * 4
  float* rsc     = (float*)(ws + 256 + 8192);   // 2048 * 4
  int*   buckets = (int*)(ws + 256 + 16384);    // 64 * 1024 * 4

  moe_zero<<<(TOKENS * OUT_W + 255) / 256, 256, 0, stream>>>(out, counts);
  moe_router<<<TOKENS / 8, 256, 0, stream>>>(x, noise, mixer, ncont, ridx, rsc);
  moe_scatter<<<(TOKENS * 2 + 255) / 256, 256, 0, stream>>>(ridx, counts, buckets);
  moe_expert_ffn<<<NEXP, 256, 0, stream>>>(x, w1s, b1s, w2s, b2s,
                                           counts, buckets, rsc, out);
}
